// YuyiPointUpdate_72997264163181
// MI455X (gfx1250) — matrix operand order repaired and checked by reading, not compiled, not run on hardware
//
#include <hip/hip_runtime.h>
#include <hip/hip_bf16.h>

typedef _Float16 v16h __attribute__((ext_vector_type(16)));
typedef _Float16 v8h  __attribute__((ext_vector_type(8)));
typedef float    v8f  __attribute__((ext_vector_type(8)));

static constexpr int Bc = 256;   // meta batch
static constexpr int Nc = 512;   // samples
static constexpr int Dc = 128;   // feat dim
static constexpr int C1 = 256;   // 2*D channels after concat / conv1 out
static constexpr int C2 = 128;   // conv2 out
static constexpr int ROWS_PER_RED_BLK = 64;

// ---------------------------------------------------------------- prep -----
__global__ void k_prep_weights(const float* __restrict__ W1,
                               const float* __restrict__ W2,
                               _Float16* __restrict__ W1h,
                               _Float16* __restrict__ W2h) {
  int i = blockIdx.x * blockDim.x + threadIdx.x;       // 256*256 + 128*256
  if (i < C1 * C1) W1h[i] = (_Float16)W1[i];
  else             W2h[i - C1 * C1] = (_Float16)W2[i - C1 * C1];
}

// inverse L1 row sums (diag excluded), one wave32 per row, both edge tensors
__global__ void k_rowinv(const float* __restrict__ Ed, const float* __restrict__ Ei,
                         float* __restrict__ invd, float* __restrict__ invi) {
  int wave = (blockIdx.x * blockDim.x + threadIdx.x) >> 5;
  int lane = threadIdx.x & 31;
  int e  = wave / (Bc * Nc);
  int rr = wave % (Bc * Nc);
  const float* row = ((e == 0) ? Ed : Ei) + (size_t)rr * Nc;
  int i = rr % Nc;
  float s = 0.f;
#pragma unroll
  for (int c = 0; c < 4; ++c) {
    int j0 = c * 128 + lane * 4;
    float4 v = *(const float4*)(row + j0);
    s += (j0 + 0 == i ? 0.f : fabsf(v.x)) + (j0 + 1 == i ? 0.f : fabsf(v.y)) +
         (j0 + 2 == i ? 0.f : fabsf(v.z)) + (j0 + 3 == i ? 0.f : fabsf(v.w));
  }
#pragma unroll
  for (int m = 16; m; m >>= 1) s += __shfl_xor(s, m, 32);
  if (lane == 0) ((e == 0) ? invd : invi)[rr] = 1.0f / fmaxf(s, 1e-12f);
}

// Pt[b][d][n] = (f16) point_node[b][n][d]  -> B-fragments become contiguous
__global__ void k_transpose_pn(const float* __restrict__ PN, _Float16* __restrict__ Pt) {
  size_t idx = (size_t)blockIdx.x * blockDim.x + threadIdx.x;  // B*D*N
  int n = (int)(idx % Nc);
  size_t t = idx / Nc;
  int d = (int)(t % Dc);
  int b = (int)(t / Dc);
  Pt[idx] = (_Float16)PN[((size_t)b * Nc + n) * Dc + d];
}

// --------------------------------------------- edge aggregation (WMMA) -----
// wave -> 16 rows (i) x 128 cols (d); K loop over N in steps of 32.
// A = normalized edge tile (scale + diag folded in), B = Pt tile.
__global__ void __launch_bounds__(128)
k_edge_gemm(const float* __restrict__ Ed, const float* __restrict__ Ei,
            const float* __restrict__ invd, const float* __restrict__ invi,
            const _Float16* __restrict__ Pt, _Float16* __restrict__ xcat) {
  int wave = (blockIdx.x * blockDim.x + threadIdx.x) >> 5;
  int lane = threadIdx.x & 31;
  int e   = wave / (Bc * (Nc / 16));
  int rem = wave % (Bc * (Nc / 16));
  int b   = rem / (Nc / 16);
  int i0  = (rem % (Nc / 16)) * 16;
  const float* E   = ((e == 0) ? Ed : Ei) + (size_t)b * Nc * Nc;
  const float* inv = ((e == 0) ? invd : invi) + (size_t)b * Nc;

  int lh = lane & 15, sel = lane >> 4;
  int i = i0 + lh;
  float r = inv[i];
  const float* rowp = E + (size_t)i * Nc;
  const _Float16* ptb = Pt + (size_t)b * Dc * Nc;

  v8f acc[8] = {};
  for (int k0 = 0; k0 < Nc; k0 += 32) {
    __builtin_prefetch(rowp + k0 + 64, 0, 1);
    // Preload all 8 B fragments (one load clause, partial waits later).
    v16h bf[8];
#pragma unroll
    for (int dd = 0; dd < 8; ++dd)
      bf[dd] = *(const v16h*)(ptb + (size_t)(dd * 16 + lh) * Nc + k0 + sel * 16);

    // A fragment: halves 0-7 -> K=k0+sel*16.., halves 8-15 -> K=k0+sel*16+8..
    int base0 = k0 + sel * 16;
    int base1 = k0 + sel * 16 + 8;
    union { float4 q[4]; float f[16]; } u;
    u.q[0] = *(const float4*)(rowp + base0);
    u.q[1] = *(const float4*)(rowp + base0 + 4);
    u.q[2] = *(const float4*)(rowp + base1);
    u.q[3] = *(const float4*)(rowp + base1 + 4);
    v16h a;
#pragma unroll
    for (int j = 0; j < 16; ++j) {
      int k = (j < 8) ? (base0 + j) : (base1 + j - 8);
      // diag mask + L1 normalization folded into one multiplier; the load
      // result is always consumed -> no divergent conditional load.
      float m = (k == i) ? 0.f : r;
      a[j] = (_Float16)(u.f[j] * m);
    }
#pragma unroll
    for (int dd = 0; dd < 8; ++dd)
      acc[dd] = __builtin_amdgcn_wmma_f32_16x16x32_f16(
          false, a, false, bf[dd], (short)0, acc[dd], false, false);
  }
  // D layout: M = vv + sel*8, N(col) = lh ; write concat (B,N,256) f16
#pragma unroll
  for (int dd = 0; dd < 8; ++dd)
#pragma unroll
    for (int vv = 0; vv < 8; ++vv) {
      int row = i0 + vv + sel * 8;
      xcat[((size_t)b * Nc + row) * C1 + e * Dc + dd * 16 + lh] = (_Float16)acc[dd][vv];
    }
}

// ------------------------------------------------- 1x1 conv GEMM (WMMA) ----
// Y[b,n,o] = sum_c Wh[o,c] * X[b,n,c];  K = 256.  wave -> 16(o) x 64(n).
__global__ void __launch_bounds__(128)
k_conv_gemm(const _Float16* __restrict__ Wh, const _Float16* __restrict__ X,
            float* __restrict__ Y, int O) {
  int wave = (blockIdx.x * blockDim.x + threadIdx.x) >> 5;
  int lane = threadIdx.x & 31;
  int ot_n = O / 16, nt_n = Nc / 64;
  int b   = wave / (ot_n * nt_n);
  int rem = wave % (ot_n * nt_n);
  int o0  = (rem / nt_n) * 16;
  int n0  = (rem % nt_n) * 64;
  int lh = lane & 15, sel = lane >> 4;
  const _Float16* wrow = Wh + (size_t)(o0 + lh) * C1;

  v8f acc[4] = {};
  for (int k0 = 0; k0 < C1; k0 += 32) {
    // Preload the 4 B fragments first, then the A chunks.
    v16h bf[4];
#pragma unroll
    for (int jt = 0; jt < 4; ++jt) {
      int n = n0 + jt * 16 + lh;
      bf[jt] = *(const v16h*)(X + ((size_t)b * Nc + n) * C1 + k0 + sel * 16);
    }
    v8h lo = *(const v8h*)(wrow + k0 + sel * 16);
    v8h hi = *(const v8h*)(wrow + k0 + sel * 16 + 8);
    v16h a;
#pragma unroll
    for (int j = 0; j < 8; ++j) { a[j] = lo[j]; a[j + 8] = hi[j]; }
#pragma unroll
    for (int jt = 0; jt < 4; ++jt)
      acc[jt] = __builtin_amdgcn_wmma_f32_16x16x32_f16(
          false, a, false, bf[jt], (short)0, acc[jt], false, false);
  }
#pragma unroll
  for (int jt = 0; jt < 4; ++jt)
#pragma unroll
    for (int vv = 0; vv < 8; ++vv) {
      int n = n0 + jt * 16 + lh;
      int o = o0 + vv + sel * 8;
      Y[((size_t)b * Nc + n) * O + o] = acc[jt][vv];
    }
}

// ----------------------------------------- BatchNorm stats (deterministic) -
__global__ void k_reduce_stats(const float* __restrict__ src,
                               float* __restrict__ psum, float* __restrict__ psq, int C) {
  int c = threadIdx.x;                       // blockDim.x == C
  size_t row0 = (size_t)blockIdx.x * ROWS_PER_RED_BLK;
  float s = 0.f, s2 = 0.f;
  for (int r = 0; r < ROWS_PER_RED_BLK; ++r) {
    float v = src[(row0 + r) * C + c];
    s += v; s2 += v * v;
  }
  psum[(size_t)blockIdx.x * C + c] = s;
  psq [(size_t)blockIdx.x * C + c] = s2;
}

__global__ void k_finalize_stats(const float* __restrict__ psum, const float* __restrict__ psq,
                                 const float* __restrict__ g, const float* __restrict__ bb,
                                 float* __restrict__ scale, float* __restrict__ bias,
                                 int C, int nblk) {
  int c = threadIdx.x;
  float s = 0.f, s2 = 0.f;
  for (int i = 0; i < nblk; ++i) { s += psum[(size_t)i * C + c]; s2 += psq[(size_t)i * C + c]; }
  const float M = (float)Bc * (float)Nc;
  float mean = s / M;
  float var  = s2 / M - mean * mean;
  float iv   = rsqrtf(var + 1e-5f);
  float sc   = g[c] * iv;
  scale[c] = sc;
  bias[c]  = bb[c] - mean * sc;
}

__global__ void k_bn_act_f16(const float* __restrict__ src, const float* __restrict__ scale,
                             const float* __restrict__ bias, _Float16* __restrict__ dst) {
  size_t idx = (size_t)blockIdx.x * blockDim.x + threadIdx.x;   // (B*N*256)
  int c = (int)(idx & (C1 - 1));
  float y = src[idx] * scale[c] + bias[c];
  dst[idx] = (_Float16)(y >= 0.f ? y : 0.01f * y);
}

__global__ void k_bn_act_out(const float* __restrict__ src, const float* __restrict__ scale,
                             const float* __restrict__ bias, float* __restrict__ dst) {
  size_t idx = (size_t)blockIdx.x * blockDim.x + threadIdx.x;   // (B*N*128)
  int c = (int)(idx & (C2 - 1));
  float y = src[idx] * scale[c] + bias[c];
  dst[idx] = (y >= 0.f ? y : 0.01f * y);
}

// ------------------------------------------------------------------ launch -
extern "C" void kernel_launch(void* const* d_in, const int* in_sizes, int n_in,
                              void* d_out, int out_size, void* d_ws, size_t ws_size,
                              hipStream_t stream) {
  const float* dist = (const float*)d_in[0];
  const float* inst = (const float*)d_in[1];
  const float* pn   = (const float*)d_in[2];
  const float* W1   = (const float*)d_in[3];
  const float* g1   = (const float*)d_in[4];
  const float* b1   = (const float*)d_in[5];
  const float* W2   = (const float*)d_in[6];
  const float* g2   = (const float*)d_in[7];
  const float* b2   = (const float*)d_in[8];
  float* out = (float*)d_out;

  char* ws = (char*)d_ws;
  size_t off = 0;
  auto take = [&](size_t bytes) -> char* {
    char* p = ws + off;
    off += (bytes + 255) & ~(size_t)255;
    return p;
  };
  float*    invd  = (float*)   take((size_t)Bc * Nc * 4);
  float*    invi  = (float*)   take((size_t)Bc * Nc * 4);
  _Float16* Pt    = (_Float16*)take((size_t)Bc * Dc * Nc * 2);
  _Float16* W1h   = (_Float16*)take((size_t)C1 * C1 * 2);
  _Float16* W2h   = (_Float16*)take((size_t)C2 * C1 * 2);
  _Float16* xcat  = (_Float16*)take((size_t)Bc * Nc * C1 * 2);
  float*    h1raw = (float*)   take((size_t)Bc * Nc * C1 * 4);
  _Float16* h1act = (_Float16*)take((size_t)Bc * Nc * C1 * 2);
  float*    h2raw = (float*)   take((size_t)Bc * Nc * C2 * 4);
  const int nredblk = (Bc * Nc) / ROWS_PER_RED_BLK;            // 2048
  float* psum1  = (float*)take((size_t)nredblk * C1 * 4);
  float* psq1   = (float*)take((size_t)nredblk * C1 * 4);
  float* psum2  = (float*)take((size_t)nredblk * C2 * 4);
  float* psq2   = (float*)take((size_t)nredblk * C2 * 4);
  float* scale1 = (float*)take(C1 * 4);
  float* bias1  = (float*)take(C1 * 4);
  float* scale2 = (float*)take(C2 * 4);
  float* bias2  = (float*)take(C2 * 4);
  (void)ws_size; (void)in_sizes; (void)n_in; (void)out_size;

  // prep
  k_prep_weights<<<(C1 * C1 + C2 * C1) / 256, 256, 0, stream>>>(W1, W2, W1h, W2h);
  k_rowinv<<<(2 * Bc * Nc * 32) / 256, 256, 0, stream>>>(dist, inst, invd, invi);
  k_transpose_pn<<<((size_t)Bc * Dc * Nc) / 256, 256, 0, stream>>>(pn, Pt);

  // edge aggregation -> concat (B,N,256) f16
  k_edge_gemm<<<(2 * Bc * (Nc / 16)) / 4, 128, 0, stream>>>(dist, inst, invd, invi, Pt, xcat);

  // conv1 -> BN1 + LeakyReLU -> f16
  k_conv_gemm<<<(Bc * (C1 / 16) * (Nc / 64)) / 4, 128, 0, stream>>>(W1h, xcat, h1raw, C1);
  k_reduce_stats<<<nredblk, C1, 0, stream>>>(h1raw, psum1, psq1, C1);
  k_finalize_stats<<<1, C1, 0, stream>>>(psum1, psq1, g1, b1, scale1, bias1, C1, nredblk);
  k_bn_act_f16<<<((size_t)Bc * Nc * C1) / 256, 256, 0, stream>>>(h1raw, scale1, bias1, h1act);

  // conv2 -> BN2 + LeakyReLU -> output f32 (B,N,128)
  k_conv_gemm<<<(Bc * (C2 / 16) * (Nc / 64)) / 4, 128, 0, stream>>>(W2h, h1act, h2raw, C2);
  k_reduce_stats<<<nredblk, C2, 0, stream>>>(h2raw, psum2, psq2, C2);
  k_finalize_stats<<<1, C2, 0, stream>>>(psum2, psq2, g2, b2, scale2, bias2, C2, nredblk);
  k_bn_act_out<<<((size_t)Bc * Nc * C2) / 256, 256, 0, stream>>>(h2raw, scale2, bias2, out);
}
